// MatryoshkaConstrativeLoss_63144609185952
// MI455X (gfx1250) — matrix operand order repaired and checked by reading, not compiled, not run on hardware
//
#include <hip/hip_runtime.h>

// ---------------------------------------------------------------------------
// Matryoshka contrastive loss, fused for MI455X (gfx1250, wave32, WMMA).
//   loss = sum_d mean_i [ logsumexp_j(cos_d(i,j)/T) - cos_d(i,i)/T ]
// cos_d uses prefix dims d in {64,128,256,512,768}. Since cos<=1, logits<=20,
// so softmax uses a FIXED max of 20 -> partial sums of exp are additive and
// can be split across blocks with atomics.
// ---------------------------------------------------------------------------

typedef unsigned short ushort_t;
typedef __attribute__((ext_vector_type(16))) __bf16 v16bf;
typedef __attribute__((ext_vector_type(8)))  float  v8f;
typedef __attribute__((ext_vector_type(4)))  int    v4i_t;

#define N_ROWS 8192
#define DFULL  768
#define NDIMS  5
#define INV_T  20.0f   // 1/temperature
#define MAXL   20.0f   // fixed softmax max (cos<=1 -> logit<=INV_T)

union FragBF { uint4 q[2]; v16bf v; };

__device__ __forceinline__ ushort_t f2bf(float f) {
  unsigned int u = __float_as_uint(f);
  u += 0x7FFFu + ((u >> 16) & 1u);      // round-to-nearest-even
  return (ushort_t)(u >> 16);
}

// Global -> LDS copy of 16 bytes per lane. Prefer CDNA5 async direct-to-LDS
// (ASYNCcnt path); fall back to load+store if the builtin is unavailable.
__device__ __forceinline__ void cp16_g2l(const ushort_t* g, ushort_t* l) {
#if defined(__gfx1250__) && __has_builtin(__builtin_amdgcn_global_load_async_to_lds_b128)
  __builtin_amdgcn_global_load_async_to_lds_b128(
      (__attribute__((address_space(1))) v4i_t*)(void*)(g),
      (__attribute__((address_space(3))) v4i_t*)(void*)(l), 0, 0);
#else
  *reinterpret_cast<uint4*>(l) = *reinterpret_cast<const uint4*>(g);
#endif
}

__device__ __forceinline__ void wait_async_lds() {
#if defined(__gfx1250__) && __has_builtin(__builtin_amdgcn_s_wait_asynccnt)
  __builtin_amdgcn_s_wait_asynccnt(0);
#endif
}

// ---------------------------------------------------------------------------
// init: zero the global row-sum accumulator and the output scalar.
// ---------------------------------------------------------------------------
__global__ void init_kernel(float* __restrict__ Gsum, float* __restrict__ out) {
  int t = blockIdx.x * 256 + threadIdx.x;
  if (t < N_ROWS * NDIMS) Gsum[t] = 0.0f;
  if (t == 0) out[0] = 0.0f;
}

// ---------------------------------------------------------------------------
// prep: one wave per row (e1 rows then e2 rows). Convert f32 -> bf16 and
// compute prefix inverse norms at the 5 breakpoints.
// ---------------------------------------------------------------------------
__global__ __launch_bounds__(256) void prep_kernel(
    const float* __restrict__ e1, const float* __restrict__ e2,
    ushort_t* __restrict__ bf1, ushort_t* __restrict__ bf2,
    float* __restrict__ inv1, float* __restrict__ inv2) {
  int wid  = (int)((blockIdx.x * blockDim.x + threadIdx.x) >> 5);
  int lane = threadIdx.x & 31;
  const float* src; ushort_t* dst; float* invp;
  if (wid < N_ROWS) {
    src = e1 + (size_t)wid * DFULL; dst = bf1 + (size_t)wid * DFULL;
    invp = inv1 + (size_t)wid * NDIMS;
  } else {
    int r = wid - N_ROWS;
    src = e2 + (size_t)r * DFULL; dst = bf2 + (size_t)r * DFULL;
    invp = inv2 + (size_t)r * NDIMS;
  }
  float s0 = 0.f, s1 = 0.f, s2 = 0.f, s3 = 0.f, s4 = 0.f;
#pragma unroll
  for (int u = 0; u < 6; ++u) {               // 6 float4 chunks per lane
    int c = lane + 32 * u;                    // chunk of elements [4c,4c+4)
    float4 f = reinterpret_cast<const float4*>(src)[c];
    float q = f.x * f.x + f.y * f.y + f.z * f.z + f.w * f.w;
    if (c < 16) s0 += q; else if (c < 32) s1 += q; else if (c < 64) s2 += q;
    else if (c < 128) s3 += q; else s4 += q;
    ushort4 h4; h4.x = f2bf(f.x); h4.y = f2bf(f.y); h4.z = f2bf(f.z); h4.w = f2bf(f.w);
    reinterpret_cast<ushort4*>(dst)[c] = h4;
  }
#pragma unroll
  for (int m = 1; m < 32; m <<= 1) {
    s0 += __shfl_xor(s0, m, 32); s1 += __shfl_xor(s1, m, 32);
    s2 += __shfl_xor(s2, m, 32); s3 += __shfl_xor(s3, m, 32);
    s4 += __shfl_xor(s4, m, 32);
  }
  if (lane == 0) {
    float c0 = s0, c1 = c0 + s1, c2 = c1 + s2, c3 = c2 + s3, c4 = c3 + s4;
    invp[0] = 1.0f / fmaxf(sqrtf(c0), 1e-8f);
    invp[1] = 1.0f / fmaxf(sqrtf(c1), 1e-8f);
    invp[2] = 1.0f / fmaxf(sqrtf(c2), 1e-8f);
    invp[3] = 1.0f / fmaxf(sqrtf(c3), 1e-8f);
    invp[4] = 1.0f / fmaxf(sqrtf(c4), 1e-8f);
  }
}

// ---------------------------------------------------------------------------
// diag: exact f32 diagonal logits; one wave per row. out -= sum_d logit_ii / N
// ---------------------------------------------------------------------------
__global__ __launch_bounds__(256) void diag_kernel(
    const float* __restrict__ e1, const float* __restrict__ e2,
    const float* __restrict__ inv1, const float* __restrict__ inv2,
    float* __restrict__ out) {
  int wid  = (int)((blockIdx.x * blockDim.x + threadIdx.x) >> 5);  // row
  int lane = threadIdx.x & 31;
  const float* a = e1 + (size_t)wid * DFULL;
  const float* b = e2 + (size_t)wid * DFULL;
  float s0 = 0.f, s1 = 0.f, s2 = 0.f, s3 = 0.f, s4 = 0.f;
#pragma unroll
  for (int u = 0; u < 6; ++u) {
    int c = lane + 32 * u;
    float4 fa = reinterpret_cast<const float4*>(a)[c];
    float4 fb = reinterpret_cast<const float4*>(b)[c];
    float q = fa.x * fb.x + fa.y * fb.y + fa.z * fb.z + fa.w * fb.w;
    if (c < 16) s0 += q; else if (c < 32) s1 += q; else if (c < 64) s2 += q;
    else if (c < 128) s3 += q; else s4 += q;
  }
#pragma unroll
  for (int m = 1; m < 32; m <<= 1) {
    s0 += __shfl_xor(s0, m, 32); s1 += __shfl_xor(s1, m, 32);
    s2 += __shfl_xor(s2, m, 32); s3 += __shfl_xor(s3, m, 32);
    s4 += __shfl_xor(s4, m, 32);
  }
  if (lane == 0) {
    float c0 = s0, c1 = c0 + s1, c2 = c1 + s2, c3 = c2 + s3, c4 = c3 + s4;
    const float* i1 = inv1 + (size_t)wid * NDIMS;
    const float* i2 = inv2 + (size_t)wid * NDIMS;
    float t = c0 * i1[0] * i2[0] + c1 * i1[1] * i2[1] + c2 * i1[2] * i2[2] +
              c3 * i1[3] * i2[3] + c4 * i1[4] * i2[4];
    atomicAdd(out, -t * INV_T * (1.0f / (float)N_ROWS));
  }
}

// ---------------------------------------------------------------------------
// loss: fused bf16 WMMA GEMM with prefix snapshots + exp accumulation.
// Block = 256 threads = 8 waves, each wave owns a 16-row A strip (LDS-resident,
// K=768 full). Grid = (64 row-blocks) x (8 column slices of 1024 cols).
// Per 32-column tile: 24 K-tiles of 32, with exp-snapshots after K=64/128/
// 256/512/768. Row sums of exp(logit-20) are atomically merged into Gsum.
// ---------------------------------------------------------------------------
__global__ __launch_bounds__(256) void loss_kernel(
    const ushort_t* __restrict__ bf1, const ushort_t* __restrict__ bf2,
    const float* __restrict__ inv1, const float* __restrict__ inv2,
    float* __restrict__ Gsum) {
  __shared__ __align__(16) ushort_t Abuf[128 * DFULL];  // 192 KB
  __shared__ __align__(16) ushort_t Bbuf[32 * DFULL];   // 48 KB
  const int tid   = threadIdx.x;
  const int lane  = tid & 31;
  const int wave  = tid >> 5;
  const int l15   = lane & 15;
  const int hh    = lane >> 4;        // lane half (WMMA fragment striping)
  const int row0  = (int)blockIdx.x * 128;
  const int wrow0 = row0 + wave * 16;

  // Stage the 128x768 bf16 A panel (rows are contiguous -> flat copy).
#pragma unroll
  for (int it = 0; it < 48; ++it) {
    int i16 = tid + 256 * it;
    cp16_g2l(bf1 + (size_t)row0 * DFULL + (size_t)i16 * 8, &Abuf[(size_t)i16 * 8]);
  }

  // Per-lane row scale factors: C-layout rows are M = hh*8 + v.
  float inv1v[NDIMS][8];
#pragma unroll
  for (int d = 0; d < NDIMS; ++d)
#pragma unroll
    for (int v = 0; v < 8; ++v)
      inv1v[d][v] = inv1[(size_t)(wrow0 + hh * 8 + v) * NDIMS + d];

  float sum[NDIMS][8];
#pragma unroll
  for (int d = 0; d < NDIMS; ++d)
#pragma unroll
    for (int v = 0; v < 8; ++v) sum[d][v] = 0.0f;

// One K-tile: A frag (16x32 bf16) from this wave's strip, B frags for the two
// 16-column halves. A and B use the same contiguous k order (k = 16*hh + s):
//  A: lanes 0..15 hold K 0..15, lanes 16..31 hold K 16..31            (M=l15)
//  B: lanes 0..15 hold K 0..15, lanes 16..31 hold K 16..31            (N=l15)
#define KT(ktv) do {                                                          \
    FragBF fa, fb;                                                            \
    { const ushort_t* pa = &Abuf[(size_t)(wave * 16 + l15) * DFULL            \
                                 + (ktv) * 32 + hh * 16];                     \
      fa.q[0] = *reinterpret_cast<const uint4*>(pa);                          \
      fa.q[1] = *reinterpret_cast<const uint4*>(pa + 8); }                    \
    { const ushort_t* pb = &Bbuf[(size_t)l15 * DFULL + (ktv) * 32 + hh * 16]; \
      fb.q[0] = *reinterpret_cast<const uint4*>(pb);                          \
      fb.q[1] = *reinterpret_cast<const uint4*>(pb + 8); }                    \
    c0 = __builtin_amdgcn_wmma_f32_16x16x32_bf16(false, fa.v, false, fb.v,    \
                                                 (short)0, c0, false, false); \
    { const ushort_t* pb = &Bbuf[(size_t)(16 + l15) * DFULL                   \
                                 + (ktv) * 32 + hh * 16];                     \
      fb.q[0] = *reinterpret_cast<const uint4*>(pb);                          \
      fb.q[1] = *reinterpret_cast<const uint4*>(pb + 8); }                    \
    c1 = __builtin_amdgcn_wmma_f32_16x16x32_bf16(false, fa.v, false, fb.v,    \
                                                 (short)0, c1, false, false); \
  } while (0)

#define SNAP(DI) do {                                                         \
    _Pragma("unroll")                                                         \
    for (int v = 0; v < 8; ++v) {                                             \
      float l0 = c0[v] * inv1v[DI][v] * inv2v[0][DI] * INV_T;                 \
      float l1 = c1[v] * inv1v[DI][v] * inv2v[1][DI] * INV_T;                 \
      sum[DI][v] += __expf(l0 - MAXL) + __expf(l1 - MAXL);                    \
    }                                                                         \
  } while (0)

#pragma unroll 1
  for (int nt = 0; nt < 32; ++nt) {
    const int cb = ((int)blockIdx.y * 32 + nt) * 32;   // 32-column tile base
    // Cooperative async stage of the 32x768 bf16 B tile.
#pragma unroll
    for (int it = 0; it < 12; ++it) {
      int i16 = tid + 256 * it;
      cp16_g2l(bf2 + (size_t)cb * DFULL + (size_t)i16 * 8, &Bbuf[(size_t)i16 * 8]);
    }
    wait_async_lds();
    __syncthreads();

    float inv2v[2][NDIMS];
#pragma unroll
    for (int h = 0; h < 2; ++h)
#pragma unroll
      for (int d = 0; d < NDIMS; ++d)
        inv2v[h][d] = inv2[(size_t)(cb + h * 16 + l15) * NDIMS + d];

    v8f c0 = {0.f, 0.f, 0.f, 0.f, 0.f, 0.f, 0.f, 0.f};
    v8f c1 = {0.f, 0.f, 0.f, 0.f, 0.f, 0.f, 0.f, 0.f};

    KT(0); KT(1); SNAP(0);                    // K = 64
    KT(2); KT(3); SNAP(1);                    // K = 128
#pragma unroll
    for (int kt = 4; kt < 8; ++kt) KT(kt);
    SNAP(2);                                  // K = 256
#pragma unroll
    for (int kt = 8; kt < 16; ++kt) KT(kt);
    SNAP(3);                                  // K = 512
#pragma unroll
    for (int kt = 16; kt < 24; ++kt) KT(kt);
    SNAP(4);                                  // K = 768

    __syncthreads();                          // protect Bbuf before next stage
  }

  // Reduce row sums across the 16 lanes of each half, then merge globally.
#pragma unroll
  for (int d = 0; d < NDIMS; ++d) {
#pragma unroll
    for (int v = 0; v < 8; ++v) {
      float s = sum[d][v];
      s += __shfl_xor(s, 1, 32);
      s += __shfl_xor(s, 2, 32);
      s += __shfl_xor(s, 4, 32);
      s += __shfl_xor(s, 8, 32);
      if (l15 == 0)
        atomicAdd(&Gsum[(size_t)(wrow0 + hh * 8 + v) * NDIMS + d], s);
    }
  }
#undef KT
#undef SNAP
}

// ---------------------------------------------------------------------------
// finalize: out += sum over (row, dim) of (20 + log Gsum) / N
// ---------------------------------------------------------------------------
__global__ __launch_bounds__(256) void finalize_kernel(
    const float* __restrict__ Gsum, float* __restrict__ out) {
  int t = blockIdx.x * 256 + threadIdx.x;
  float v = 0.0f;
  if (t < N_ROWS * NDIMS) v = MAXL + logf(Gsum[t]);
#pragma unroll
  for (int m = 1; m < 32; m <<= 1) v += __shfl_xor(v, m, 32);
  if ((threadIdx.x & 31) == 0) atomicAdd(out, v * (1.0f / (float)N_ROWS));
}

// ---------------------------------------------------------------------------
extern "C" void kernel_launch(void* const* d_in, const int* in_sizes, int n_in,
                              void* d_out, int out_size, void* d_ws, size_t ws_size,
                              hipStream_t stream) {
  const float* e1 = (const float*)d_in[0];
  const float* e2 = (const float*)d_in[1];
  float* out = (float*)d_out;

  // Workspace layout (~25.7 MB).
  ushort_t* bf1 = (ushort_t*)d_ws;
  ushort_t* bf2 = bf1 + (size_t)N_ROWS * DFULL;
  float* inv1 = (float*)(bf2 + (size_t)N_ROWS * DFULL);
  float* inv2 = inv1 + (size_t)N_ROWS * NDIMS;
  float* Gsum = inv2 + (size_t)N_ROWS * NDIMS;

  init_kernel<<<160, 256, 0, stream>>>(Gsum, out);
  prep_kernel<<<2 * N_ROWS / 8, 256, 0, stream>>>(e1, e2, bf1, bf2, inv1, inv2);
  diag_kernel<<<N_ROWS / 8, 256, 0, stream>>>(e1, e2, inv1, inv2, out);
  loss_kernel<<<dim3(N_ROWS / 128, 8, 1), 256, 0, stream>>>(bf1, bf2, inv1, inv2, Gsum);
  finalize_kernel<<<160, 256, 0, stream>>>(Gsum, out);
}
